// MAGIC_73194832658522
// MI455X (gfx1250) — matrix operand order repaired and checked by reading, not compiled, not run on hardware
//
#include <hip/hip_runtime.h>

typedef _Float16 v16h __attribute__((ext_vector_type(16)));
typedef float    v8f  __attribute__((ext_vector_type(8)));
typedef int      v4i_ __attribute__((vector_size(16)));

#define AS1 __attribute__((address_space(1)))
#define AS3 __attribute__((address_space(3)))

#if defined(__has_builtin)
#  if __has_builtin(__builtin_amdgcn_global_load_async_to_lds_b128)
#    define HAVE_ASYNC_LDS 1
#  endif
#  if __has_builtin(__builtin_amdgcn_s_wait_asynccnt)
#    define HAVE_WAIT_ASYNC 1
#  endif
#endif

// 16-byte chunk copy global -> LDS. Async (ASYNCcnt) on CDNA5, VGPR fallback else.
__device__ __forceinline__ void copy16_async(const void* g, void* l) {
#ifdef HAVE_ASYNC_LDS
  __builtin_amdgcn_global_load_async_to_lds_b128(
      (AS1 v4i_*)g, (AS3 v4i_*)l, 0, 0);
#else
  *(v4i_*)l = *(const v4i_*)g;
#endif
}
__device__ __forceinline__ void async_wait() {
#ifdef HAVE_ASYNC_LDS
#  ifdef HAVE_WAIT_ASYNC
  __builtin_amdgcn_s_wait_asynccnt(0);
#  else
  asm volatile("s_wait_asynccnt 0x0" ::: "memory");
#  endif
#endif
}

// ---------------------------------------------------------------------------
// WMMA GEMM with async-LDS staging:
//   C[M,N](f32) = A[M,K](f16) @ B[K,N](f16) (+C if acc) (+bias) (act)
// grid = (N/64, M/16), block = 128 (4 waves; wave w owns cols [w*16, w*16+16))
// Requires N % 64 == 0, K % 32 == 0, 16B-aligned A/B, lda/ldb % 8 == 0.
// flags: bit0 = accumulate into existing C, bits[2:1]: 0=none 1=relu 2=elu
// ---------------------------------------------------------------------------
__global__ __launch_bounds__(128) void k_gemm16(
    const _Float16* __restrict__ A, int lda,
    const _Float16* __restrict__ B, int ldb,
    float* __restrict__ C, int ldc,
    const float* __restrict__ bias,
    int N, int K, int flags) {
  __shared__ alignas(16) _Float16 sA[16 * 32];  // 1 KB: rows x k
  __shared__ alignas(16) _Float16 sB[32 * 64];  // 4 KB: k x cols
  int t = threadIdx.x;
  int wave = t >> 5, lane = t & 31, tm = lane & 15, hi = lane >> 4;
  int row0   = blockIdx.y << 4;
  int colblk = blockIdx.x << 6;
  int col0   = colblk + (wave << 4);

  v8f acc = {};
  if (flags & 1) {
#pragma unroll
    for (int r = 0; r < 8; ++r)
      acc[r] = C[(size_t)(row0 + hi * 8 + r) * ldc + col0 + tm];
  }

  for (int k0 = 0; k0 < K; k0 += 32) {
    __syncthreads();  // previous tile fully consumed by all waves
    if (t < 64) {     // A tile: 16 rows x 64B = 64 chunks
      int r = t >> 2, seg = t & 3;
      copy16_async(A + (size_t)(row0 + r) * lda + k0 + seg * 8,
                   sA + r * 32 + seg * 8);
    }
#pragma unroll
    for (int it = 0; it < 2; ++it) {  // B tile: 32 rows x 128B = 256 chunks
      int id = t + it * 128;
      int r = id >> 3, seg = id & 7;
      copy16_async(B + (size_t)(k0 + r) * ldb + colblk + seg * 8,
                   sB + r * 64 + seg * 8);
    }
    async_wait();
    __syncthreads();

    v16h af, bf;
#pragma unroll
    for (int v = 0; v < 8; ++v) {
      int ka = ((v >= 4) ? 16 : 0) + hi * 8 + (v & 3) * 2;
      af[2 * v]     = sA[tm * 32 + ka];
      af[2 * v + 1] = sA[tm * 32 + ka + 1];
      int kb = ((v >= 4) ? 16 : 0) + hi * 8 + (v & 3) * 2;
      int cl = (wave << 4) + tm;
      bf[2 * v]     = sB[kb * 64 + cl];
      bf[2 * v + 1] = sB[(kb + 1) * 64 + cl];
    }
    acc = __builtin_amdgcn_wmma_f32_16x16x32_f16(false, af, false, bf,
                                                 (short)0, acc, false, false);
  }

  int act = (flags >> 1) & 3;
  if (col0 < N) {
#pragma unroll
    for (int r = 0; r < 8; ++r) {
      int row = row0 + hi * 8 + r;
      int col = col0 + tm;
      float val = acc[r];
      if (bias) val += bias[col];
      if (act == 1)      val = val > 0.f ? val : 0.f;
      else if (act == 2) val = val > 0.f ? val : (__expf(val) - 1.f);
      C[(size_t)row * ldc + col] = val;
    }
  }
}

__global__ void k_cast(const float* __restrict__ src, _Float16* __restrict__ dst, int n) {
  int i = blockIdx.x * blockDim.x + threadIdx.x;
  if (i < n) dst[i] = (_Float16)src[i];
}

// LSTM cell elementwise (torch gate order i,f,g,o), grid 256 x block 256
__global__ void k_lstm(const float* __restrict__ gates, const float* __restrict__ cell,
                       float* __restrict__ h_out, float* __restrict__ c_out,
                       _Float16* __restrict__ h16) {
  int idx = blockIdx.x * blockDim.x + threadIdx.x;  // 65536
  int n = idx >> 8, c = idx & 255;
  const float* g = gates + (size_t)n * 1024;
  float gi = g[c], gf = g[256 + c], gg = g[512 + c], go = g[768 + c];
  float si = 1.f / (1.f + __expf(-gi));
  float sf = 1.f / (1.f + __expf(-gf));
  float so = 1.f / (1.f + __expf(-go));
  float cn = sf * cell[idx] + si * tanhf(gg);
  float hn = so * tanhf(cn);
  c_out[idx] = cn;
  h_out[idx] = hn;
  h16[idx] = (_Float16)hn;
}

// ----------------------- JAX threefry2x32 / gumbel -------------------------
__device__ __forceinline__ void tf_round(unsigned& x0, unsigned& x1, int r) {
  x0 += x1;
  x1 = (x1 << r) | (x1 >> (32 - r));
  x1 ^= x0;
}
__device__ __forceinline__ void threefry2x32(unsigned k0, unsigned k1,
                                             unsigned c0, unsigned c1,
                                             unsigned& o0, unsigned& o1) {
  unsigned k2 = k0 ^ k1 ^ 0x1BD11BDAu;
  unsigned x0 = c0 + k0, x1 = c1 + k1;
  tf_round(x0, x1, 13); tf_round(x0, x1, 15); tf_round(x0, x1, 26); tf_round(x0, x1, 6);
  x0 += k1; x1 += k2 + 1u;
  tf_round(x0, x1, 17); tf_round(x0, x1, 29); tf_round(x0, x1, 16); tf_round(x0, x1, 24);
  x0 += k2; x1 += k0 + 2u;
  tf_round(x0, x1, 13); tf_round(x0, x1, 15); tf_round(x0, x1, 26); tf_round(x0, x1, 6);
  x0 += k0; x1 += k1 + 3u;
  tf_round(x0, x1, 17); tf_round(x0, x1, 29); tf_round(x0, x1, 16); tf_round(x0, x1, 24);
  x0 += k1; x1 += k2 + 4u;
  tf_round(x0, x1, 13); tf_round(x0, x1, 15); tf_round(x0, x1, 26); tf_round(x0, x1, 6);
  x0 += k2; x1 += k0 + 5u;
  o0 = x0; o1 = x1;
}
// flat index fi over shape (256,256,2); total 131072, JAX split-counter layout
__device__ __forceinline__ float jax_gumbel(unsigned key1, unsigned fi) {
  const unsigned half = 65536u;
  unsigned o0, o1, bits;
  if (fi < half) { threefry2x32(0u, key1, fi, fi + half, o0, o1); bits = o0; }
  else           { threefry2x32(0u, key1, fi - half, fi, o0, o1); bits = o1; }
  float f = __uint_as_float((bits >> 9) | 0x3f800000u) - 1.0f;  // [0,1)
  float u = 1e-10f + f * (1.0f - 1e-10f);
  return -logf(-logf(u));
}

// ---------------------------------------------------------------------------
// Fused scheduler: row i (block), all j: z1=relu(A[i]+B[j]+b1) [128],
// z2=relu(z1@w2+b2) [32] via WMMA, logits=z2@w3+b3, hard-gumbel argmax -> adj.
// grid = 256 blocks, block = 256 threads (8 waves; wave handles 32 j's)
// ---------------------------------------------------------------------------
__global__ __launch_bounds__(256) void k_sched(
    const float* __restrict__ Af, const float* __restrict__ Bf,
    const float* __restrict__ b1,
    const _Float16* __restrict__ w2,
    const float* __restrict__ b2,
    const float* __restrict__ w3, const float* __restrict__ b3,
    float* __restrict__ adj, unsigned key1) {
  __shared__ float    sA[128];
  __shared__ float    sb1[128];
  __shared__ alignas(16) _Float16 sw2[128 * 32];   // 8 KB, async-staged
  __shared__ float    sz2[256][32];
  __shared__ float    sw3[64];
  __shared__ float    sb3v[2];
  __shared__ float    sb2v[32];
  int t = threadIdx.x;
  int i = blockIdx.x;
  // stage w2 (8 KB contiguous) via async b128: 512 chunks, 2 per thread
#pragma unroll
  for (int it = 0; it < 2; ++it) {
    int id = t + it * 256;
    copy16_async(w2 + id * 8, sw2 + id * 8);
  }
  if (t < 128) { sA[t] = Af[i * 128 + t]; sb1[t] = b1[t]; }
  if (t < 64) sw3[t] = w3[t];
  if (t < 32) sb2v[t] = b2[t];
  if (t < 2)  sb3v[t] = b3[t];
  async_wait();
  __syncthreads();

  int wave = t >> 5, lane = t & 31, tm = lane & 15, hi = lane >> 4;
  for (int jt = 0; jt < 2; ++jt) {
    int jbase = wave * 32 + jt * 16;
    int jrow  = jbase + tm;
    const float* Brow = Bf + (size_t)jrow * 128;
    v8f acc0 = {}, acc1 = {};
    for (int kk = 0; kk < 4; ++kk) {
      v16h af, bf0, bf1;
#pragma unroll
      for (int v = 0; v < 8; ++v) {
        int ka = kk * 32 + ((v >= 4) ? 16 : 0) + hi * 8 + (v & 3) * 2;
        float z0 = sA[ka] + Brow[ka] + sb1[ka];
        float z1 = sA[ka + 1] + Brow[ka + 1] + sb1[ka + 1];
        af[2 * v]     = (_Float16)(z0 > 0.f ? z0 : 0.f);
        af[2 * v + 1] = (_Float16)(z1 > 0.f ? z1 : 0.f);
        int kb = kk * 32 + ((v >= 4) ? 16 : 0) + hi * 8 + (v & 3) * 2;
        bf0[2 * v]     = sw2[kb * 32 + tm];
        bf0[2 * v + 1] = sw2[(kb + 1) * 32 + tm];
        bf1[2 * v]     = sw2[kb * 32 + 16 + tm];
        bf1[2 * v + 1] = sw2[(kb + 1) * 32 + 16 + tm];
      }
      acc0 = __builtin_amdgcn_wmma_f32_16x16x32_f16(false, af, false, bf0,
                                                    (short)0, acc0, false, false);
      acc1 = __builtin_amdgcn_wmma_f32_16x16x32_f16(false, af, false, bf1,
                                                    (short)0, acc1, false, false);
    }
#pragma unroll
    for (int r = 0; r < 8; ++r) {
      int j = jbase + hi * 8 + r;
      float v0 = acc0[r] + sb2v[tm];
      float v1 = acc1[r] + sb2v[16 + tm];
      sz2[j][tm]      = v0 > 0.f ? v0 : 0.f;
      sz2[j][16 + tm] = v1 > 0.f ? v1 : 0.f;
    }
  }
  __syncthreads();

  float l0 = sb3v[0], l1 = sb3v[1];
  for (int k = 0; k < 32; ++k) {
    float z = sz2[t][k];
    l0 += z * sw3[k * 2];
    l1 += z * sw3[k * 2 + 1];
  }
  unsigned fi = ((unsigned)(i * 256 + t)) * 2u;
  float g0 = jax_gumbel(key1, fi);
  float g1 = jax_gumbel(key1, fi + 1u);
  adj[i * 256 + t] = ((l1 + g1) > (l0 + g0)) ? 1.0f : 0.0f;
}

// ci[n,h] = dot(h[n, h*F:(h+1)*F], a[h])
__global__ void k_attcoef(const float* __restrict__ h, const float* __restrict__ a,
                          float* __restrict__ out, int heads, int F) {
  int idx = blockIdx.x * blockDim.x + threadIdx.x;
  if (idx >= 256 * heads) return;
  int n = idx / heads, hd = idx % heads;
  const float* hp = h + (size_t)n * heads * F + (size_t)hd * F;
  const float* ap = a + (size_t)hd * F;
  float s = 0.f;
  for (int f = 0; f < F; ++f) s += hp[f] * ap[f];
  out[idx] = s;
}

// per-(i,head) masked softmax over j; forced self loop; write f16 attention
__global__ void k_attsoftmax(const float* __restrict__ ci, const float* __restrict__ cj,
                             const float* __restrict__ adj, _Float16* __restrict__ att,
                             int heads) {
  __shared__ float red[256];
  int head = blockIdx.x % heads;
  int i = blockIdx.x / heads;
  int j = threadIdx.x;
  float ad = (i == j) ? 1.0f : adj[i * 256 + j];
  float e = ci[i * heads + head] + cj[j * heads + head];
  e = e > 0.f ? e : 0.2f * e;  // leaky relu 0.2
  float val = ad > 0.f ? e : -9e15f;
  red[j] = val; __syncthreads();
  for (int s = 128; s > 0; s >>= 1) { if (j < s) red[j] = fmaxf(red[j], red[j + s]); __syncthreads(); }
  float m = red[0]; __syncthreads();
  float p = ad > 0.f ? __expf(e - m) : 0.f;
  red[j] = p; __syncthreads();
  for (int s = 128; s > 0; s >>= 1) { if (j < s) red[j] += red[j + s]; __syncthreads(); }
  float ssum = red[0];
  att[(size_t)head * 65536 + i * 256 + j] = (_Float16)(p / ssum);
}

__global__ void k_value(const float* __restrict__ h, const float* __restrict__ comm,
                        const float* __restrict__ w_v, const float* __restrict__ b_v,
                        float* __restrict__ out) {
  __shared__ float red[256];
  int n = blockIdx.x, t = threadIdx.x;
  float p = h[n * 256 + t] * w_v[t] + comm[n * 256 + t] * w_v[256 + t];
  red[t] = p; __syncthreads();
  for (int s = 128; s > 0; s >>= 1) { if (t < s) red[t] += red[t + s]; __syncthreads(); }
  if (t == 0) out[n] = red[0] + b_v[0];
}

__global__ void k_action(const float* __restrict__ h, const float* __restrict__ comm,
                         const float* __restrict__ w_a, const float* __restrict__ b_a,
                         float* __restrict__ out) {
  __shared__ float lg[10];
  __shared__ float lse_s;
  int n = blockIdx.x, t = threadIdx.x;
  if (t < 10) {
    float s = b_a[t];
    for (int k = 0; k < 256; ++k) s += h[n * 256 + k] * w_a[k * 10 + t];
    for (int k = 0; k < 256; ++k) s += comm[n * 256 + k] * w_a[(256 + k) * 10 + t];
    lg[t] = s;
  }
  __syncthreads();
  if (t == 0) {
    float m = lg[0];
    for (int a2 = 1; a2 < 10; ++a2) m = fmaxf(m, lg[a2]);
    float s = 0.f;
    for (int a2 = 0; a2 < 10; ++a2) s += __expf(lg[a2] - m);
    lse_s = m + __logf(s);
  }
  __syncthreads();
  if (t < 10) out[n * 10 + t] = lg[t] - lse_s;
}

// ---------------------------------------------------------------------------
extern "C" void kernel_launch(void* const* d_in, const int* in_sizes, int n_in,
                              void* d_out, int out_size, void* d_ws, size_t ws_size,
                              hipStream_t stream) {
  (void)in_sizes; (void)n_in; (void)out_size; (void)ws_size;
  const float* obs   = (const float*)d_in[0];
  const float* h_in  = (const float*)d_in[1];
  const float* c_in  = (const float*)d_in[2];
  const float* w_obs = (const float*)d_in[3];  const float* b_obs = (const float*)d_in[4];
  const float* w_ih  = (const float*)d_in[5];  const float* b_ih  = (const float*)d_in[6];
  const float* w_hh  = (const float*)d_in[7];  const float* b_hh  = (const float*)d_in[8];
  const float* w_me  = (const float*)d_in[9];  const float* b_me  = (const float*)d_in[10];
  const float* w_md  = (const float*)d_in[11]; const float* b_md  = (const float*)d_in[12];
  const float* s1_w1 = (const float*)d_in[13]; const float* s1_b1 = (const float*)d_in[14];
  const float* s1_w2 = (const float*)d_in[15]; const float* s1_b2 = (const float*)d_in[16];
  const float* s1_w3 = (const float*)d_in[17]; const float* s1_b3 = (const float*)d_in[18];
  const float* s2_w1 = (const float*)d_in[19]; const float* s2_b1 = (const float*)d_in[20];
  const float* s2_w2 = (const float*)d_in[21]; const float* s2_b2 = (const float*)d_in[22];
  const float* s2_w3 = (const float*)d_in[23]; const float* s2_b3 = (const float*)d_in[24];
  const float* g1_w  = (const float*)d_in[25]; const float* g1_ai = (const float*)d_in[26];
  const float* g1_aj = (const float*)d_in[27]; const float* g1_b  = (const float*)d_in[28];
  const float* g2_w  = (const float*)d_in[29]; const float* g2_ai = (const float*)d_in[30];
  const float* g2_aj = (const float*)d_in[31]; const float* g2_b  = (const float*)d_in[32];
  const float* w_v   = (const float*)d_in[33]; const float* b_v   = (const float*)d_in[34];
  const float* w_a   = (const float*)d_in[35]; const float* b_a   = (const float*)d_in[36];

  float* out_action = (float*)d_out;      // [1,256,10]
  float* out_value  = out_action + 2560;  // [256,1]
  float* out_h      = out_value + 256;    // [256,256]
  float* out_c      = out_h + 65536;      // [256,256]

  // --- bump allocator in d_ws (total ~8.6 MB) ---
  char* base = (char*)d_ws;
  size_t off = 0;
  auto alloc = [&](size_t bytes) -> void* {
    off = (off + 255) & ~(size_t)255;
    void* p = base + off;
    off += bytes;
    return p;
  };
  // f16 buffers
  _Float16* obs16   = (_Float16*)alloc(32768 * 2);
  _Float16* wobs16  = (_Float16*)alloc(32768 * 2);
  _Float16* x16     = (_Float16*)alloc(65536 * 2);
  _Float16* hprev16 = (_Float16*)alloc(65536 * 2);
  _Float16* wih16   = (_Float16*)alloc(262144 * 2);
  _Float16* whh16   = (_Float16*)alloc(262144 * 2);
  _Float16* hnew16  = (_Float16*)alloc(65536 * 2);
  _Float16* wme16   = (_Float16*)alloc(65536 * 2);
  _Float16* comm16  = (_Float16*)alloc(65536 * 2);
  _Float16* s1w116  = (_Float16*)alloc(65536 * 2);
  _Float16* s2w116  = (_Float16*)alloc(65536 * 2);
  _Float16* s1w216  = (_Float16*)alloc(4096 * 2);
  _Float16* s2w216  = (_Float16*)alloc(4096 * 2);
  _Float16* g1w16   = (_Float16*)alloc(131072 * 2);
  _Float16* h116    = (_Float16*)alloc(131072 * 2);
  _Float16* att116  = (_Float16*)alloc(262144 * 2);
  _Float16* comm216 = (_Float16*)alloc(131072 * 2);
  _Float16* g2w16   = (_Float16*)alloc(131072 * 2);
  _Float16* h216    = (_Float16*)alloc(65536 * 2);
  _Float16* att216  = (_Float16*)alloc(65536 * 2);
  _Float16* gat2o16 = (_Float16*)alloc(65536 * 2);
  _Float16* wmd16   = (_Float16*)alloc(65536 * 2);
  // f32 buffers
  float* xf    = (float*)alloc(65536 * 4);
  float* gates = (float*)alloc(262144 * 4);
  float* commf0= (float*)alloc(65536 * 4);
  float* A1f   = (float*)alloc(32768 * 4);
  float* B1f   = (float*)alloc(32768 * 4);
  float* A2f   = (float*)alloc(32768 * 4);
  float* B2f   = (float*)alloc(32768 * 4);
  float* adj1  = (float*)alloc(65536 * 4);
  float* adj2  = (float*)alloc(65536 * 4);
  float* h1f   = (float*)alloc(131072 * 4);
  float* ci1   = (float*)alloc(1024 * 4);
  float* cj1   = (float*)alloc(1024 * 4);
  float* ci2   = (float*)alloc(256 * 4);
  float* cj2   = (float*)alloc(256 * 4);
  float* comm2f= (float*)alloc(131072 * 4);
  float* h2f   = (float*)alloc(65536 * 4);
  float* gat2f = (float*)alloc(65536 * 4);
  float* commF = (float*)alloc(65536 * 4);

  auto cast = [&](const float* s, _Float16* d, int n) {
    k_cast<<<(n + 255) / 256, 256, 0, stream>>>(s, d, n);
  };

  // --- casts of inputs/weights ---
  cast(obs, obs16, 32768);
  cast(w_obs, wobs16, 32768);
  cast(h_in, hprev16, 65536);
  cast(w_ih, wih16, 262144);
  cast(w_hh, whh16, 262144);
  cast(w_me, wme16, 65536);
  cast(s1_w1, s1w116, 65536);
  cast(s2_w1, s2w116, 65536);
  cast(s1_w2, s1w216, 4096);
  cast(s2_w2, s2w216, 4096);
  cast(g1_w, g1w16, 131072);
  cast(g2_w, g2w16, 131072);
  cast(w_md, wmd16, 65536);

  dim3 blk(128);
  // x = obs @ w_obs + b_obs    [256,128]x[128,256]
  k_gemm16<<<dim3(4, 16), blk, 0, stream>>>(obs16, 128, wobs16, 256, xf, 256, b_obs, 256, 128, 0);
  cast(xf, x16, 65536);
  // gates = x@w_ih + b_ih + h@w_hh + b_hh   [256,256]x[256,1024] x2
  k_gemm16<<<dim3(16, 16), blk, 0, stream>>>(x16, 256, wih16, 1024, gates, 1024, b_ih, 1024, 256, 0);
  k_gemm16<<<dim3(16, 16), blk, 0, stream>>>(hprev16, 256, whh16, 1024, gates, 1024, b_hh, 1024, 256, 1);
  // LSTM elementwise -> h_new, c_new (direct to d_out) + h_new f16
  k_lstm<<<256, 256, 0, stream>>>(gates, c_in, out_h, out_c, hnew16);
  // comm = h_new @ w_me + b_me
  k_gemm16<<<dim3(4, 16), blk, 0, stream>>>(hnew16, 256, wme16, 256, commf0, 256, b_me, 256, 256, 0);
  cast(commf0, comm16, 65536);

  // scheduler 1: A = comm@w1_top, B = comm@w1_bot (w1 is [512,128])
  k_gemm16<<<dim3(2, 16), blk, 0, stream>>>(comm16, 256, s1w116, 128, A1f, 128, nullptr, 128, 256, 0);
  k_gemm16<<<dim3(2, 16), blk, 0, stream>>>(comm16, 256, s1w116 + 256 * 128, 128, B1f, 128, nullptr, 128, 256, 0);
  k_sched<<<256, 256, 0, stream>>>(A1f, B1f, s1_b1, s1w216, s1_b2, s1_w3, s1_b3, adj1, 1u);
  // scheduler 2 (on comm_ori, key(2))
  k_gemm16<<<dim3(2, 16), blk, 0, stream>>>(comm16, 256, s2w116, 128, A2f, 128, nullptr, 128, 256, 0);
  k_gemm16<<<dim3(2, 16), blk, 0, stream>>>(comm16, 256, s2w116 + 256 * 128, 128, B2f, 128, nullptr, 128, 256, 0);
  k_sched<<<256, 256, 0, stream>>>(A2f, B2f, s2_b1, s2w216, s2_b2, s2_w3, s2_b3, adj2, 2u);

  // GAT1: h1 = comm @ g1_w   [256,256]x[256,512] -> [256,4,128]
  k_gemm16<<<dim3(8, 16), blk, 0, stream>>>(comm16, 256, g1w16, 512, h1f, 512, nullptr, 512, 256, 0);
  cast(h1f, h116, 131072);
  k_attcoef<<<4, 256, 0, stream>>>(h1f, g1_ai, ci1, 4, 128);
  k_attcoef<<<4, 256, 0, stream>>>(h1f, g1_aj, cj1, 4, 128);
  k_attsoftmax<<<1024, 256, 0, stream>>>(ci1, cj1, adj1, att116, 4);
  for (int head = 0; head < 4; ++head) {
    // out_h = att_h @ h1_h, +g1_b slice, ELU fused  -> comm2[:, head*128:...]
    k_gemm16<<<dim3(2, 16), blk, 0, stream>>>(att116 + head * 65536, 256,
                                              h116 + head * 128, 512,
                                              comm2f + head * 128, 512,
                                              g1_b + head * 128, 128, 256, 4 /*elu*/);
  }
  cast(comm2f, comm216, 131072);

  // GAT2: h2 = comm2 @ g2_w   [256,512]x[512,256]
  k_gemm16<<<dim3(4, 16), blk, 0, stream>>>(comm216, 512, g2w16, 256, h2f, 256, nullptr, 256, 512, 0);
  cast(h2f, h216, 65536);
  k_attcoef<<<1, 256, 0, stream>>>(h2f, g2_ai, ci2, 1, 256);
  k_attcoef<<<1, 256, 0, stream>>>(h2f, g2_aj, cj2, 1, 256);
  k_attsoftmax<<<256, 256, 0, stream>>>(ci2, cj2, adj2, att216, 1);
  // gat2out = att2 @ h2 + g2_b
  k_gemm16<<<dim3(4, 16), blk, 0, stream>>>(att216, 256, h216, 256, gat2f, 256, g2_b, 256, 256, 0);
  cast(gat2f, gat2o16, 65536);
  // comm_final = gat2out @ w_md + b_md
  k_gemm16<<<dim3(4, 16), blk, 0, stream>>>(gat2o16, 256, wmd16, 256, commF, 256, b_md, 256, 256, 0);

  // heads
  k_value<<<256, 256, 0, stream>>>(out_h, commF, w_v, b_v, out_value);
  k_action<<<256, 32, 0, stream>>>(out_h, commF, w_a, b_a, out_action);
}
